// CodeSynthesisModel_70970039599563
// MI455X (gfx1250) — compile-verified, matrix operand order checked by reading, not run on hardware
//
#include <hip/hip_runtime.h>

typedef __attribute__((ext_vector_type(16))) _Float16 v16h;
typedef __attribute__((ext_vector_type(8)))  float    v8f;

#define TPB   256
#define CHUNK 512   // nodes per block in main kernel (2 per thread)

// ---------------------------------------------------------------------------
// Wave32 reduction (ds/permlane path on gfx1250)
// ---------------------------------------------------------------------------
__device__ __forceinline__ float wave_red32(float x) {
#pragma unroll
  for (int o = 16; o > 0; o >>= 1) x += __shfl_down(x, o, 32);
  return x;
}

// ---------------------------------------------------------------------------
// Kernel 1: precompute PE table, fused attention vector v = Wa1@Wa2,
//           c = ba1@Wa2 + ba2, last[b][152], C_b, and zero accumulators.
// One block, trivial cost.
// ---------------------------------------------------------------------------
__global__ void prep_kernel(const int* __restrict__ trees,
                            const float* __restrict__ lstm,
                            const float* __restrict__ first,
                            const float* __restrict__ emb,
                            const float* __restrict__ Wa1,
                            const float* __restrict__ ba1,
                            const float* __restrict__ Wa2,
                            const float* __restrict__ ba2,
                            float* __restrict__ pe,    // 200*8
                            float* __restrict__ v,     // 304
                            float* __restrict__ cb,    // 1
                            float* __restrict__ last,  // B*152
                            float* __restrict__ Cb,    // B
                            float* __restrict__ accz,  // B*(24+64+1) contiguous zero region
                            int B, int N) {
  const int tid = threadIdx.x;

  // PE table: pe[pos, d]; freq exponent uses (d & ~1)
  for (int i = tid; i < 200 * 8; i += TPB) {
    int pos = i >> 3, d = i & 7;
    float div = __expf((float)(d & ~1) * (-logf(10000.0f) / 8.0f));
    float x = (float)pos * div;
    pe[i] = (d & 1) ? __cosf(x) : __sinf(x);
  }
  // Fused v[j] = sum_k Wa1[j,k] * Wa2[k]
  for (int j = tid; j < 304; j += TPB) {
    float s = 0.f;
    for (int k = 0; k < 152; ++k) s += Wa1[j * 152 + k] * Wa2[k];
    v[j] = s;
  }
  if (tid == 0) {
    float s = ba2[0];
    for (int k = 0; k < 152; ++k) s += ba1[k] * Wa2[k];
    cb[0] = s;
  }
  // Zero the accumulators (must happen every launch: graph replays)
  for (int i = tid; i < B * 89; i += TPB) accz[i] = 0.f;
  __syncthreads();

  // last[b] = node_vec[b, N-1, :]
  for (int i = tid; i < B * 152; i += TPB) {
    int b = i / 152, d = i - b * 152;
    const int* t = trees + (size_t)(b * N + (N - 1)) * 4;
    float val;
    if      (d <  8) val = pe[t[0] * 8 + d];
    else if (d < 16) val = pe[t[1] * 8 + (d - 8)];
    else if (d < 24) val = emb[t[2] * 8 + (d - 16)];
    else if (d < 88) val = lstm[((size_t)b * N + t[3]) * 64 + (d - 24)];
    else             val = first[b * 64 + (d - 88)];
    last[i] = val;
  }
  __syncthreads();

  // C_b = c + last[b]*v[0:152] + first[b]*v[152+88 : 304]
  for (int b = tid; b < B; b += TPB) {
    float s = cb[0];
    for (int d = 0; d < 152; ++d) s += last[b * 152 + d] * v[d];
    for (int j = 0; j < 64; ++j)  s += first[b * 64 + j] * v[152 + 88 + j];
    Cb[b] = s;
  }
}

// ---------------------------------------------------------------------------
// Kernel 2: streaming gather + fused attention + weighted reduction.
// grid = (ceil(N/CHUNK), B). Per node: att = C_b + pe/emb.v2a + lstm.v2b,
// then accumulate att_sum (24 pe/emb dims + 64 lstm dims + scalar S for the
// constant `first` slice). Wave-level shuffle reduce, then f32 global atomics.
// ---------------------------------------------------------------------------
__global__ void main_kernel(const int* __restrict__ trees,
                            const float* __restrict__ lstm,
                            const float* __restrict__ first,
                            const float* __restrict__ emb_g,
                            const float* __restrict__ pe_g,
                            const float* __restrict__ v_g,     // 304; node part at +152
                            const float* __restrict__ Cb_g,
                            float* __restrict__ acc24,         // B*24
                            float* __restrict__ accL,          // B*64
                            float* __restrict__ Sacc,          // B
                            int B, int N) {
  __shared__ float pe_s[1600];
  __shared__ float emb_s[1600];
  __shared__ float v2_s[88];   // 24 pe/emb + 64 lstm weights
  __shared__ float Cb_s;

  const int tid = threadIdx.x;
  const int b = blockIdx.y;

  for (int i = tid; i < 1600; i += TPB) { pe_s[i] = pe_g[i]; emb_s[i] = emb_g[i]; }
  for (int i = tid; i < 88; i += TPB)   v2_s[i] = v_g[152 + i];
  if (tid == 0) Cb_s = Cb_g[b];
  __syncthreads();

  float a24[24];
  float aL[64];
#pragma unroll
  for (int j = 0; j < 24; ++j) a24[j] = 0.f;
#pragma unroll
  for (int j = 0; j < 64; ++j) aL[j] = 0.f;
  float S = 0.f;

  const int n0 = blockIdx.x * CHUNK;
  const int nEnd = (n0 + CHUNK < N) ? (n0 + CHUNK) : N;

  for (int n = n0 + tid; n < nEnd; n += TPB) {
    const int4 tt = ((const int4*)trees)[(size_t)b * N + n];
    float nv[24];
    float att = Cb_s;
#pragma unroll
    for (int j = 0; j < 8; ++j) { float p = pe_s[tt.x * 8 + j];  nv[j]      = p; att += p * v2_s[j]; }
#pragma unroll
    for (int j = 0; j < 8; ++j) { float p = pe_s[tt.y * 8 + j];  nv[8 + j]  = p; att += p * v2_s[8 + j]; }
#pragma unroll
    for (int j = 0; j < 8; ++j) { float p = emb_s[tt.z * 8 + j]; nv[16 + j] = p; att += p * v2_s[16 + j]; }

    const float4* lr = (const float4*)(lstm + ((size_t)b * N + tt.w) * 64);
    float4 L[16];
#pragma unroll
    for (int i = 0; i < 16; ++i) {
      L[i] = lr[i];
      att += L[i].x * v2_s[24 + 4 * i] + L[i].y * v2_s[25 + 4 * i] +
             L[i].z * v2_s[26 + 4 * i] + L[i].w * v2_s[27 + 4 * i];
    }
    S += att;
#pragma unroll
    for (int j = 0; j < 24; ++j) a24[j] += att * nv[j];
#pragma unroll
    for (int i = 0; i < 16; ++i) {
      aL[4 * i + 0] += att * L[i].x;
      aL[4 * i + 1] += att * L[i].y;
      aL[4 * i + 2] += att * L[i].z;
      aL[4 * i + 3] += att * L[i].w;
    }
  }

  // Cross-lane reduce, lane 0 commits with global f32 atomics.
  const int lane = tid & 31;
  S = wave_red32(S);
  if (lane == 0) atomicAdd(&Sacc[b], S);
#pragma unroll
  for (int j = 0; j < 24; ++j) {
    float r = wave_red32(a24[j]);
    if (lane == 0) atomicAdd(&acc24[b * 24 + j], r);
  }
#pragma unroll
  for (int j = 0; j < 64; ++j) {
    float r = wave_red32(aL[j]);
    if (lane == 0) atomicAdd(&accL[b * 64 + j], r);
  }
}

// ---------------------------------------------------------------------------
// Kernel 3: MLP tail. GEMM1 (128x152 @ 152x32) runs on the matrix pipe via
// v_wmma_f32_16x16x32_f16 (A/B staged as f16 in LDS, K padded to 160).
// Remaining <=64x32 matmuls are scalar. One block, 8 waves.
// ---------------------------------------------------------------------------
__global__ void tail_kernel(const float* __restrict__ first,
                            const float* __restrict__ W1,  const float* __restrict__ b1,
                            const float* __restrict__ W2,  const float* __restrict__ b2,
                            const float* __restrict__ Wf1, const float* __restrict__ bf1,
                            const float* __restrict__ Wf2, const float* __restrict__ bf2,
                            const float* __restrict__ Wt1, const float* __restrict__ bt1,
                            const float* __restrict__ Wt2, const float* __restrict__ bt2,
                            const float* __restrict__ last,
                            const float* __restrict__ acc24,
                            const float* __restrict__ accL,
                            const float* __restrict__ Sacc,
                            float* __restrict__ out, int B) {
  __shared__ __align__(16) char smem[59392];
  _Float16* Xs  = (_Float16*)smem;      // 128*160 f16 = 40960 B
  _Float16* Wsm = Xs + 128 * 160;       // 160*32  f16 = 10240 B
  _Float16* H1  = Wsm + 160 * 32;       // 128*32  f16 =  8192 B

  const int tid = threadIdx.x;
  const int M = 2 * B;                  // 128 rows of hidden_in

  // Build X = hidden_in (row 2b = last[b], row 2b+1 = att_sum[b]); pad K to 160.
  for (int i = tid; i < M * 160; i += TPB) {
    int r = i / 160, k = i - r * 160;
    int b = r >> 1, s = r & 1;
    float val = 0.f;
    if (k < 152) {
      if (s == 0) val = last[b * 152 + k];
      else        val = (k < 24) ? acc24[b * 24 + k]
                   : (k < 88)    ? accL[b * 64 + (k - 24)]
                                 : first[b * 64 + (k - 88)] * Sacc[b];
    }
    Xs[i] = (_Float16)val;
  }
  // W1 (152x32) -> f16, K-padded
  for (int i = tid; i < 160 * 32; i += TPB) {
    int k = i >> 5, nn = i & 31;
    Wsm[i] = (_Float16)((k < 152) ? W1[k * 32 + nn] : 0.f);
  }
  __syncthreads();

  // GEMM1 on WMMA: 8 waves x (2 n-tiles) x (5 k-steps)
  const int wv = tid >> 5, lane = tid & 31;
  const int l2 = lane & 15, hi = lane >> 4;
  if (wv < M / 16) {
    const int bm = wv * 16;
    for (int nt = 0; nt < 2; ++nt) {
      v8f acc = {};
      for (int kt = 0; kt < 5; ++kt) {
        const int bk = kt * 32;
        v16h af, bf;
#pragma unroll
        for (int j = 0; j < 16; ++j) {
          // 16-bit A layout: K = (j&7) + ((j>>3)<<4) + (lane_half<<3)
          int kk = (j & 7) + ((j >> 3) << 4) + (hi << 3);
          af[j] = Xs[(bm + l2) * 160 + bk + kk];
          bf[j] = Wsm[(bk + kk) * 32 + nt * 16 + l2];
        }
        acc = __builtin_amdgcn_wmma_f32_16x16x32_f16(
            /*neg_a=*/false, af, /*neg_b=*/false, bf,
            /*c_mod=*/(short)0, acc, /*reuse_a=*/false, /*reuse_b=*/false);
      }
      // bias + relu, store f16 per C layout (VGPR i -> M = hi*8+i, N = l2)
#pragma unroll
      for (int i = 0; i < 8; ++i) {
        int m = bm + hi * 8 + i;
        int nn = nt * 16 + l2;
        float vv = acc[i] + b1[nn];
        H1[m * 32 + nn] = (_Float16)(vv > 0.f ? vv : 0.f);
      }
    }
  }
  __syncthreads();

  // Reuse front of smem for f32 phase buffers (doesn't overlap H1 @ 51200)
  float* Hh = (float*)smem;        // 64*32
  float* S3 = Hh + 64 * 32;        // 64*32
  float* S4 = S3 + 64 * 32;        // 64*16
  float* T5 = S4 + 64 * 16;        // 64*16

  // GEMM2 + relu + reshape (B,2,16)->(B,32)
  for (int o = tid; o < M * 16; o += TPB) {
    int r = o >> 4, j = o & 15;
    float s = b2[j];
    for (int k = 0; k < 32; ++k) s += (float)H1[r * 32 + k] * W2[k * 16 + j];
    s = s > 0.f ? s : 0.f;
    int b = r >> 1, c = (r & 1) * 16 + j;
    Hh[b * 32 + c] = s;
  }
  __syncthreads();
  // GEMM3 + relu
  for (int o = tid; o < B * 32; o += TPB) {
    int b = o >> 5, j = o & 31;
    float s = bf1[j];
    for (int k = 0; k < 32; ++k) s += Hh[b * 32 + k] * Wf1[k * 32 + j];
    S3[o] = s > 0.f ? s : 0.f;
  }
  __syncthreads();
  // GEMM4 + relu
  for (int o = tid; o < B * 16; o += TPB) {
    int b = o >> 4, j = o & 15;
    float s = bf2[j];
    for (int k = 0; k < 32; ++k) s += S3[b * 32 + k] * Wf2[k * 16 + j];
    S4[o] = s > 0.f ? s : 0.f;
  }
  __syncthreads();
  // GEMM5 (no relu)
  for (int o = tid; o < B * 16; o += TPB) {
    int b = o >> 4, j = o & 15;
    float s = bt1[j];
    for (int k = 0; k < 16; ++k) s += S4[b * 16 + k] * Wt1[k * 16 + j];
    T5[o] = s;
  }
  __syncthreads();
  // GEMM6 -> score
  for (int b = tid; b < B; b += TPB) {
    float s = bt2[0];
    for (int k = 0; k < 16; ++k) s += T5[b * 16 + k] * Wt2[k];
    out[b] = s;
  }
}

// ---------------------------------------------------------------------------
extern "C" void kernel_launch(void* const* d_in, const int* in_sizes, int n_in,
                              void* d_out, int out_size, void* d_ws, size_t ws_size,
                              hipStream_t stream) {
  const int*   trees = (const int*)  d_in[0];
  const float* lstm  = (const float*)d_in[1];
  const float* first = (const float*)d_in[2];
  const float* emb   = (const float*)d_in[3];
  const float* Wa1   = (const float*)d_in[4];
  const float* ba1   = (const float*)d_in[5];
  const float* Wa2   = (const float*)d_in[6];
  const float* ba2   = (const float*)d_in[7];
  const float* W1    = (const float*)d_in[8];
  const float* b1    = (const float*)d_in[9];
  const float* W2    = (const float*)d_in[10];
  const float* b2    = (const float*)d_in[11];
  const float* Wf1   = (const float*)d_in[12];
  const float* bf1   = (const float*)d_in[13];
  const float* Wf2   = (const float*)d_in[14];
  const float* bf2   = (const float*)d_in[15];
  const float* Wt1   = (const float*)d_in[16];
  const float* bt1   = (const float*)d_in[17];
  const float* Wt2   = (const float*)d_in[18];
  const float* bt2   = (const float*)d_in[19];
  float* out = (float*)d_out;

  const int B = in_sizes[2] / 64;          // first_notes: (B,64)
  const int N = in_sizes[0] / (B * 4);     // trees: (B,N,4)

  // Workspace layout (floats)
  float* ws    = (float*)d_ws;
  float* pe    = ws;                 // 1600
  float* v     = pe + 1600;          // 304
  float* cb    = v + 304;            // 1 (padded to 16)
  float* last  = cb + 16;            // B*152
  float* Cb    = last + B * 152;     // B
  float* acc24 = Cb + B;             // B*24   (start of contiguous zero region, B*89)
  float* accL  = acc24 + B * 24;     // B*64
  float* Sacc  = accL + B * 64;      // B

  prep_kernel<<<1, TPB, 0, stream>>>(trees, lstm, first, emb, Wa1, ba1, Wa2, ba2,
                                     pe, v, cb, last, Cb, acc24, B, N);

  dim3 grid((N + CHUNK - 1) / CHUNK, B);
  main_kernel<<<grid, TPB, 0, stream>>>(trees, lstm, first, emb, pe, v, Cb,
                                        acc24, accL, Sacc, B, N);

  tail_kernel<<<1, TPB, 0, stream>>>(first, W1, b1, W2, b2, Wf1, bf1, Wf2, bf2,
                                     Wt1, bt1, Wt2, bt2, last, acc24, accL, Sacc,
                                     out, B);
}
